// GiPHEmbedding_NE_49701361549770
// MI455X (gfx1250) — matrix operand order repaired and checked by reading, not compiled, not run on hardware
//
#include <hip/hip_runtime.h>

typedef __attribute__((ext_vector_type(16))) __bf16 v16bf;
typedef __attribute__((ext_vector_type(8)))  float  v8f;

#define D   128
#define P   8192
#define LVL 16
#define KE  8
#define PK  65536   // P * KE

// --- gfx1250 async load-to-LDS (guarded: arity differs across toolchains) ---
#if defined(__has_builtin)
#if __has_builtin(__builtin_amdgcn_global_load_async_to_lds_b128) && \
    __has_builtin(__builtin_amdgcn_s_wait_asynccnt)
#define HAVE_ASYNC_LDS 1
#endif
#endif
#ifndef HAVE_ASYNC_LDS
#define HAVE_ASYNC_LDS 0
#endif

#if HAVE_ASYNC_LDS
typedef int v4i_g __attribute__((vector_size(16)));              // int __vector(4)
typedef __attribute__((address_space(1))) v4i_g gas_v4i;         // global AS
typedef __attribute__((address_space(3))) v4i_g las_v4i;         // LDS AS
#endif

// Stage an 8192-dword (32KB) weight tile global->LDS.
__device__ __forceinline__ void stage_w(unsigned* dstLds, const unsigned* srcGlob, int tid) {
#if HAVE_ASYNC_LDS
#pragma unroll
    for (int i = 0; i < 8; ++i) {
        int o = tid * 4 + i * 1024;       // dword index, 16B-aligned chunks
        __builtin_amdgcn_global_load_async_to_lds_b128(
            (gas_v4i*)(srcGlob + o), (las_v4i*)(dstLds + o), 0, 0);
    }
    __builtin_amdgcn_s_wait_asynccnt(0);  // our wave's LDS writes visible
#else
    for (int i = tid; i < 8192; i += 256) dstLds[i] = srcGlob[i];
#endif
}

__device__ __forceinline__ unsigned pack_bf2(float lo, float hi) {
    union { __bf16 b[2]; unsigned u; } x;
    x.b[0] = (__bf16)lo;    // fptrunc -> native bf16 cvt, RNE
    x.b[1] = (__bf16)hi;
    return x.u;
}

union Frag16 { v16bf v; unsigned u[8]; };

// A tile in LDS: row-major bf16, 64 dwords (=128 cols) per row.
// ISA 16-bit A 16x32 layout: lane m (0-15) holds row m, K={0..7,16..23};
// lane m+16 (kb=1) holds K={8..15,24..31}.
__device__ __forceinline__ v16bf load_a_frag(const unsigned* lds, int m, int kb, int kstep) {
    Frag16 f;
#pragma unroll
    for (int j = 0; j < 8; ++j) {
        int idx = 16 * kstep + ((j & 4) ? 8 : 0) + 4 * kb + (j & 3);
        f.u[j] = lds[m * 64 + idx];
    }
    return f.v;
}
// W^T in LDS: [col][row] bf16, 64 dwords per col.
// ISA 16-bit B 32x16 layout: lane n holds col n, K={0..7,16..23}; lane n+16 (kb=1) holds K={8..15,24..31}.
__device__ __forceinline__ v16bf load_b_frag(const unsigned* lds, int col, int kb, int kstep) {
    Frag16 f;
#pragma unroll
    for (int j = 0; j < 8; ++j)
        f.u[j] = lds[col * 64 + 16 * kstep + ((j & 4) ? 8 : 0) + 4 * kb + (j & 3)];
    return f.v;
}

__device__ __forceinline__ v8f wmma_bf16(v16bf a, v16bf b, v8f c) {
    return __builtin_amdgcn_wmma_f32_16x16x32_bf16(false, a, false, b, (short)0, c, false, false);
}

// ---------------------------------------------------------------------------
// Convert 6 f32 128x128 weight matrices to bf16, transposed ([col][row]).
__global__ void k_convert_weights(const float* w0, const float* w1, const float* w2,
                                  const float* w3, const float* w4, const float* w5,
                                  unsigned short* dst) {
    int id = blockIdx.x * 256 + threadIdx.x;   // 6*16384 threads
    int mat = id >> 14;
    int o   = id & 16383;
    int c = o >> 7, r = o & 127;
    const float* src;
    switch (mat) {
        case 0: src = w0; break; case 1: src = w1; break;
        case 2: src = w2; break; case 3: src = w3; break;
        case 4: src = w4; break; default: src = w5; break;
    }
    union { __bf16 b[2]; unsigned short s[2]; } cv;
    cv.b[0] = (__bf16)src[r * 128 + c];
    dst[(mat << 14) + o] = cv.s[0];
}

// ---------------------------------------------------------------------------
// h = relu(x@W1 + b1) @ W2 + b2   -> y [N x 128] f32
__global__ void __launch_bounds__(256)
k_node_transform(const float* __restrict__ x,
                 const unsigned short* __restrict__ wts,
                 const float* __restrict__ b1, const float* __restrict__ b2,
                 float* __restrict__ y) {
    __shared__ __align__(16) unsigned buf[9216];  // [0..1023] A, [1024..9215] W^T
    const int tid = threadIdx.x;
    const int wave = tid >> 5, lane = tid & 31;
    const int n = lane & 15, kb = lane >> 4;
    const int row0 = blockIdx.x * 16;

    {   // stage x tile (16 rows x 128 cols, f32 -> bf16), float4 loads
        const float4* xs = (const float4*)(x + row0 * 128);
#pragma unroll
        for (int j = 0; j < 2; ++j) {
            int idx = tid * 2 + j;             // [0,512) float4 index
            float4 v = xs[idx];
            buf[idx * 2]     = pack_bf2(v.x, v.y);
            buf[idx * 2 + 1] = pack_bf2(v.z, v.w);
        }
    }
    stage_w(buf + 1024, (const unsigned*)wts, tid);           // W1^T (mat 0)
    if (tid < 8) __builtin_prefetch((const char*)wts + 32768 + tid * 4096, 0, 3);
    __syncthreads();

    v8f acc = {0.f,0.f,0.f,0.f,0.f,0.f,0.f,0.f};
#pragma unroll
    for (int ks = 0; ks < 4; ++ks) {
        v16bf a = load_a_frag(buf, n, kb, ks);
        v16bf b = load_b_frag(buf + 1024, 16 * wave + n, kb, ks);
        acc = wmma_bf16(a, b, acc);
    }
    __syncthreads();
    {   // t = relu(acc + b1) -> bf16 in buf[0..1023]
        int col = 16 * wave + n;
        float bias = b1[col];
        unsigned short* t16 = (unsigned short*)buf;
#pragma unroll
        for (int r = 0; r < 8; ++r) {
            int m = r + 8 * kb;
            union { __bf16 b; unsigned short s; } cv;
            cv.b = (__bf16)fmaxf(acc[r] + bias, 0.f);
            t16[m * 128 + col] = cv.s;
        }
    }
    stage_w(buf + 1024, (const unsigned*)wts + 8192, tid);    // W2^T (mat 1)
    __syncthreads();

    v8f acc2 = {0.f,0.f,0.f,0.f,0.f,0.f,0.f,0.f};
#pragma unroll
    for (int ks = 0; ks < 4; ++ks) {
        v16bf a = load_a_frag(buf, n, kb, ks);
        v16bf b = load_b_frag(buf + 1024, 16 * wave + n, kb, ks);
        acc2 = wmma_bf16(a, b, acc2);
    }
    int col = 16 * wave + n;
    float bias = b2[col];
#pragma unroll
    for (int r = 0; r < 8; ++r) {
        int m = r + 8 * kb;
        y[(row0 + m) * 128 + col] = acc2[r] + bias;
    }
}

// ---------------------------------------------------------------------------
// Level 0 fwd / level L-1 bwd: y' = relu(upd_b) + y
__global__ void k_level_init(const float* __restrict__ y,
                             const float* __restrict__ f_upd_b,
                             const float* __restrict__ b_upd_b,
                             float* __restrict__ out) {
    int id = blockIdx.x * 256 + threadIdx.x;   // P*D threads
    int p = id >> 7, c = id & 127;
    out[p * 256 + c] = fmaxf(f_upd_b[c], 0.f) + y[p * 128 + c];
    int nb = (LVL - 1) * P + p;
    out[nb * 256 + 128 + c] = fmaxf(b_upd_b[c], 0.f) + y[nb * 128 + c];
}

// ---------------------------------------------------------------------------
// One forward level: gather, msg GEMM + relu + mean(8), update GEMM + residual.
// Block = 16 dst nodes (128 edges). out columns [0..127] hold hf.
__global__ void __launch_bounds__(256)
k_forward_level(const float* __restrict__ out_r, float* __restrict__ out_w,
                const int* __restrict__ src_local,
                const unsigned short* __restrict__ wts,
                const float* __restrict__ pre_b, const float* __restrict__ upd_b,
                const float* __restrict__ y, int l) {
    __shared__ __align__(16) unsigned buf[16384];  // 64KB: [0..8191] A, [8192..] W^T
    const int tid = threadIdx.x;
    const int wave = tid >> 5, lane = tid & 31;
    const int n = lane & 15, kb = lane >> 4;
    const int p0 = blockIdx.x * 16;
    const int E0 = p0 * KE;

    {   // gather 128 edge rows of yf[l-1] -> bf16 LDS (float4 loads)
        int row = tid >> 1, half = tid & 1;
        int src = src_local[(l - 1) * PK + E0 + row];
        const float4* sp = (const float4*)(out_r + ((l - 1) * P + src) * 256 + half * 64);
#pragma unroll
        for (int c = 0; c < 16; ++c) {
            float4 v = sp[c];
            buf[row * 64 + half * 32 + 2 * c]     = pack_bf2(v.x, v.y);
            buf[row * 64 + half * 32 + 2 * c + 1] = pack_bf2(v.z, v.w);
        }
    }
    stage_w(buf + 8192, (const unsigned*)wts + 2 * 8192, tid);   // f_pre_w^T
    // prefetch f_upd_w^T for the second GEMM while the first one runs
    if (tid < 8) __builtin_prefetch((const char*)wts + 3 * 32768 + tid * 4096, 0, 3);
    __syncthreads();

    // wave w owns edge-row tile w (rows 16w..16w+15), all 8 N-tiles
    v8f acc[8];
#pragma unroll
    for (int t = 0; t < 8; ++t) {
        v8f z = {0.f,0.f,0.f,0.f,0.f,0.f,0.f,0.f};
        acc[t] = z;
    }
#pragma unroll
    for (int ks = 0; ks < 4; ++ks) {
        v16bf a = load_a_frag(buf + wave * 16 * 64, n, kb, ks);
#pragma unroll
        for (int t = 0; t < 8; ++t) {
            v16bf b = load_b_frag(buf + 8192, 16 * t + n, kb, ks);
            acc[t] = wmma_bf16(a, b, acc[t]);
        }
    }
    __syncthreads();

    {   // z(dst, col) = mean_r relu(acc + bias); dst_local = 2*wave + kb
        unsigned short* z16 = (unsigned short*)buf;
        int dl = 2 * wave + kb;
#pragma unroll
        for (int t = 0; t < 8; ++t) {
            int col = 16 * t + n;
            float bias = pre_b[col];
            float s = 0.f;
#pragma unroll
            for (int r = 0; r < 8; ++r) s += fmaxf(acc[t][r] + bias, 0.f);
            union { __bf16 b; unsigned short sh; } cv;
            cv.b = (__bf16)(s * 0.125f);
            z16[dl * 128 + col] = cv.sh;
        }
    }
    stage_w(buf + 8192, (const unsigned*)wts + 3 * 8192, tid);   // f_upd_w^T
    __syncthreads();

    v8f acc2 = {0.f,0.f,0.f,0.f,0.f,0.f,0.f,0.f};
#pragma unroll
    for (int ks = 0; ks < 4; ++ks) {
        v16bf a = load_a_frag(buf, n, kb, ks);
        v16bf b = load_b_frag(buf + 8192, 16 * wave + n, kb, ks);
        acc2 = wmma_bf16(a, b, acc2);
    }
    int col = 16 * wave + n;
    float bias = upd_b[col];
#pragma unroll
    for (int r = 0; r < 8; ++r) {
        int node = l * P + p0 + r + 8 * kb;
        out_w[node * 256 + col] = fmaxf(acc2[r] + bias, 0.f) + y[node * 128 + col];
    }
}

// ---------------------------------------------------------------------------
// Reverse: m = relu(yb[l+1] @ b_pre_w + b_pre_b)  (8192 unique rows)
__global__ void __launch_bounds__(256)
k_rev_msg(const float* __restrict__ out_r,
          const unsigned short* __restrict__ wts,
          const float* __restrict__ pre_b,
          float* __restrict__ m, int l) {
    __shared__ __align__(16) unsigned buf[9216];
    const int tid = threadIdx.x;
    const int wave = tid >> 5, lane = tid & 31;
    const int n = lane & 15, kb = lane >> 4;
    const int row0 = blockIdx.x * 16;

#pragma unroll
    for (int j = 0; j < 2; ++j) {   // stage yb[l+1] rows -> bf16 (float4 loads)
        int idx = tid * 2 + j;      // [0,512) float4 index
        int row = idx >> 5, c4 = idx & 31;
        int node = (l + 1) * P + row0 + row;
        const float4* sp = (const float4*)(out_r + node * 256 + 128);
        float4 v = sp[c4];
        buf[row * 64 + c4 * 2]     = pack_bf2(v.x, v.y);
        buf[row * 64 + c4 * 2 + 1] = pack_bf2(v.z, v.w);
    }
    stage_w(buf + 1024, (const unsigned*)wts + 4 * 8192, tid);   // b_pre_w^T
    __syncthreads();

    v8f acc = {0.f,0.f,0.f,0.f,0.f,0.f,0.f,0.f};
#pragma unroll
    for (int ks = 0; ks < 4; ++ks) {
        v16bf a = load_a_frag(buf, n, kb, ks);
        v16bf b = load_b_frag(buf + 1024, 16 * wave + n, kb, ks);
        acc = wmma_bf16(a, b, acc);
    }
    int col = 16 * wave + n;
    float bias = pre_b[col];
#pragma unroll
    for (int r = 0; r < 8; ++r) {
        int row = r + 8 * kb;
        m[(row0 + row) * 128 + col] = fmaxf(acc[r] + bias, 0.f);
    }
}

__global__ void k_zero(float* zsum, float* deg) {
    int id = blockIdx.x * 256 + threadIdx.x;   // P*D threads
    zsum[id] = 0.f;
    if (id < P) deg[id] = 0.f;
}

// scatter-add msg rows (dedup: edge e uses m[e/K]) into zsum[src], count deg
__global__ void k_rev_scatter(const float* __restrict__ m,
                              const int* __restrict__ src_local,
                              float* __restrict__ zsum, float* __restrict__ deg, int l) {
    int id = blockIdx.x * 256 + threadIdx.x;   // PK*8 threads
    int e = id >> 3, q = id & 7;               // q selects 16-col chunk
    int s = src_local[l * PK + e];
    int row = e >> 3;                          // dst_local = e / K
    const float* mm = m + row * 128 + q * 16;
    float* zz = zsum + s * 128 + q * 16;
#pragma unroll
    for (int c = 0; c < 16; ++c) atomicAdd(zz + c, mm[c]);
    if (q == 0) atomicAdd(deg + s, 1.0f);
}

// yb[l] = relu((zsum/max(deg,1)) @ b_upd_w + b_upd_b) + y[l]
__global__ void __launch_bounds__(256)
k_rev_update(const float* __restrict__ zsum, const float* __restrict__ deg,
             const unsigned short* __restrict__ wts,
             const float* __restrict__ upd_b, const float* __restrict__ y,
             float* __restrict__ out_w, int l) {
    __shared__ __align__(16) unsigned buf[9216];
    const int tid = threadIdx.x;
    const int wave = tid >> 5, lane = tid & 31;
    const int n = lane & 15, kb = lane >> 4;
    const int row0 = blockIdx.x * 16;

#pragma unroll
    for (int j = 0; j < 2; ++j) {   // stage z = zsum/deg -> bf16 (float4 loads)
        int idx = tid * 2 + j;
        int row = idx >> 5, c4 = idx & 31;
        int p = row0 + row;
        float sc = 1.0f / fmaxf(deg[p], 1.0f);
        const float4* zp = (const float4*)(zsum + p * 128);
        float4 v = zp[c4];
        buf[row * 64 + c4 * 2]     = pack_bf2(v.x * sc, v.y * sc);
        buf[row * 64 + c4 * 2 + 1] = pack_bf2(v.z * sc, v.w * sc);
    }
    stage_w(buf + 1024, (const unsigned*)wts + 5 * 8192, tid);   // b_upd_w^T
    __syncthreads();

    v8f acc = {0.f,0.f,0.f,0.f,0.f,0.f,0.f,0.f};
#pragma unroll
    for (int ks = 0; ks < 4; ++ks) {
        v16bf a = load_a_frag(buf, n, kb, ks);
        v16bf b = load_b_frag(buf + 1024, 16 * wave + n, kb, ks);
        acc = wmma_bf16(a, b, acc);
    }
    int col = 16 * wave + n;
    float bias = upd_b[col];
#pragma unroll
    for (int r = 0; r < 8; ++r) {
        int node = l * P + row0 + r + 8 * kb;
        out_w[node * 256 + 128 + col] = fmaxf(acc[r] + bias, 0.f) + y[node * 128 + col];
    }
}

// ---------------------------------------------------------------------------
extern "C" void kernel_launch(void* const* d_in, const int* in_sizes, int n_in,
                              void* d_out, int out_size, void* d_ws, size_t ws_size,
                              hipStream_t stream) {
    const float* x       = (const float*)d_in[0];
    const int*   src     = (const int*)  d_in[1];
    const float* nt_w1   = (const float*)d_in[2];
    const float* nt_b1   = (const float*)d_in[3];
    const float* nt_w2   = (const float*)d_in[4];
    const float* nt_b2   = (const float*)d_in[5];
    const float* f_pre_w = (const float*)d_in[6];
    const float* f_pre_b = (const float*)d_in[7];
    const float* f_upd_w = (const float*)d_in[8];
    const float* f_upd_b = (const float*)d_in[9];
    const float* b_pre_w = (const float*)d_in[10];
    const float* b_pre_b = (const float*)d_in[11];
    const float* b_upd_w = (const float*)d_in[12];
    const float* b_upd_b = (const float*)d_in[13];
    float* out = (float*)d_out;   // [N, 256]: cols 0..127 = hf, 128..255 = hb

    char* ws = (char*)d_ws;
    float*          y    = (float*)ws;                          // 64 MB
    unsigned short* wts  = (unsigned short*)(ws + 67108864);    // 192 KB bf16 W^T x6
    float*          mbuf = (float*)(ws + 67305472);             // 4 MB
    float*          zsum = (float*)(ws + 71499776);             // 4 MB
    float*          deg  = (float*)(ws + 75694080);             // 32 KB

    k_convert_weights<<<384, 256, 0, stream>>>(nt_w1, nt_w2, f_pre_w, f_upd_w,
                                               b_pre_w, b_upd_w, wts);
    k_node_transform<<<(LVL * P) / 16, 256, 0, stream>>>(x, wts, nt_b1, nt_b2, y);
    k_level_init<<<(P * D) / 256, 256, 0, stream>>>(y, f_upd_b, b_upd_b, out);

    for (int l = 1; l < LVL; ++l)
        k_forward_level<<<P / 16, 256, 0, stream>>>(out, out, src, wts,
                                                    f_pre_b, f_upd_b, y, l);

    for (int l = LVL - 2; l >= 0; --l) {
        k_rev_msg<<<P / 16, 256, 0, stream>>>(out, wts, b_pre_b, mbuf, l);
        k_zero<<<(P * D) / 256, 256, 0, stream>>>(zsum, deg);
        k_rev_scatter<<<(PK * 8) / 256, 256, 0, stream>>>(mbuf, src, zsum, deg, l);
        k_rev_update<<<P / 16, 256, 0, stream>>>(zsum, deg, wts, b_upd_b, y, out, l);
    }
}
